// EGNNLayer_12472585027713
// MI455X (gfx1250) — matrix operand order repaired and checked by reading, not compiled, not run on hardware
//
#include <hip/hip_runtime.h>

typedef __attribute__((ext_vector_type(16))) __bf16 v16bf;
typedef __attribute__((ext_vector_type(8)))  __bf16 v8bf;
typedef __attribute__((ext_vector_type(8)))  float  v8f;

#define N_NODES 50000
#define N_EDGES 800000

__device__ __forceinline__ float silu_f(float x) {
    return x / (1.0f + __expf(-x));
}

__device__ __forceinline__ v16bf cat16(v8bf lo, v8bf hi) {
    return __builtin_shufflevector(lo, hi, 0,1,2,3,4,5,6,7,8,9,10,11,12,13,14,15);
}

__device__ __forceinline__ v8f wmma_bf16(v16bf a, v16bf b, v8f c) {
    return __builtin_amdgcn_wmma_f32_16x16x32_bf16(false, a, false, b, (short)0, c, false, false);
}

// Wave-level GEMM: 64xK A tile (LDS bf16) x Kx32 B slice (2 N-tiles from row-major
// weight W[n][k], global bf16) -> 8 accumulators acc[mt][s] = D tile (mt-th 16 rows,
// s-th 16 cols). Per K-step the two B fragments are loaded once and reused across
// 4 M-tiles (4x L2-traffic amortization), each A fragment reused across 2 N-tiles.
//
// A frag (16-bit A 16x32, ISA 7.12.2): lane m=lane&15, hi=lane>>4,
//   elems 0..7  -> K = kb + hi*8 + e ; elems 8..15 -> K = kb + 16 + hi*8 + (e-8)
// B frag: lane n=lane&15, same K order as A   (B[k][n] = W[n][k])
template<int K>
__device__ __forceinline__ void wave_gemm4x2(const __bf16* __restrict__ ldsA, int lda,
                                             const __bf16* __restrict__ W,   int ldw,
                                             int ncol0, int ncol1, int lane,
                                             v8f acc[4][2]) {
    const int mrow = lane & 15;
    const int hi   = lane >> 4;
    const __bf16* b0row = W + (size_t)ncol0 * ldw;
    const __bf16* b1row = W + (size_t)ncol1 * ldw;
    #pragma unroll
    for (int kb = 0; kb < K; kb += 32) {
        v16bf b0 = cat16(*(const v8bf*)(b0row + kb + hi * 8),
                         *(const v8bf*)(b0row + kb + 16 + hi * 8));
        v16bf b1 = cat16(*(const v8bf*)(b1row + kb + hi * 8),
                         *(const v8bf*)(b1row + kb + 16 + hi * 8));
        #pragma unroll
        for (int mt = 0; mt < 4; ++mt) {
            const __bf16* arow = ldsA + (mt * 16 + mrow) * lda;
            v16bf a = cat16(*(const v8bf*)(arow + kb + hi * 8),
                            *(const v8bf*)(arow + kb + 16 + hi * 8));
            acc[mt][0] = wmma_bf16(a, b0, acc[mt][0]);
            acc[mt][1] = wmma_bf16(a, b1, acc[mt][1]);
        }
    }
}

__device__ __forceinline__ void zero_acc(v8f acc[4][2]) {
    #pragma unroll
    for (int mt = 0; mt < 4; ++mt)
        #pragma unroll
        for (int s = 0; s < 2; ++s)
            acc[mt][s] = (v8f){0.f,0.f,0.f,0.f,0.f,0.f,0.f,0.f};
}

// ---------------- small utility kernels ----------------

__global__ void convert_pad_kernel(const float* __restrict__ src, __bf16* __restrict__ dst,
                                   int rows, int cols, int ldDst) {
    int idx = blockIdx.x * blockDim.x + threadIdx.x;
    int total = rows * ldDst;
    if (idx >= total) return;
    int r = idx / ldDst;
    int c = idx - r * ldDst;
    dst[idx] = (c < cols) ? (__bf16)src[r * cols + c] : (__bf16)0.0f;
}

__global__ void zero_kernel(float* __restrict__ p, int n) {
    int i = blockIdx.x * blockDim.x + threadIdx.x;
    if (i < n) p[i] = 0.0f;
}

__global__ void coord_out_kernel(const float* __restrict__ coord,
                                 const float* __restrict__ cAgg,
                                 const float* __restrict__ cnt,
                                 float* __restrict__ out, int n) {
    int i = blockIdx.x * blockDim.x + threadIdx.x;
    if (i >= n) return;
    float c = cnt[i];
    c = (c < 1.0f) ? 1.0f : c;
    float inv = 1.0f / c;
    out[i * 3 + 0] = coord[i * 3 + 0] + cAgg[i * 3 + 0] * inv;
    out[i * 3 + 1] = coord[i * 3 + 1] + cAgg[i * 3 + 1] * inv;
    out[i * 3 + 2] = coord[i * 3 + 2] + cAgg[i * 3 + 2] * inv;
}

// ---------------- fused edge kernel: one block = 64 edges ----------------

__global__ void __launch_bounds__(256)
egnn_edge_kernel(const float* __restrict__ h, const int* __restrict__ ei,
                 const float* __restrict__ coord,
                 const __bf16* __restrict__ We1b, const float* __restrict__ be1,
                 const __bf16* __restrict__ We2b, const float* __restrict__ be2,
                 const __bf16* __restrict__ Wc1b, const float* __restrict__ bc1,
                 const float* __restrict__ Wc2,
                 float* __restrict__ agg, float* __restrict__ cAgg,
                 float* __restrict__ cnt) {
    __shared__ __bf16 sIn[64 * 544];   // edge_in tile (radial col 512, zero pad to 544)
    __shared__ __bf16 sM [64 * 256];   // after edge-MLP layer 1
    __shared__ __bf16 sEF[64 * 256];   // edge_feat
    __shared__ int    sR[64];
    __shared__ int    sC[64];
    __shared__ float  sCD[64][3];

    const int tid = threadIdx.x;
    const int e0  = blockIdx.x * 64;

    if (tid < 64) {
        int e = e0 + tid;
        int r = ei[e];
        int c = ei[N_EDGES + e];
        sR[tid] = r; sC[tid] = c;
        float dx = coord[r * 3 + 0] - coord[c * 3 + 0];
        float dy = coord[r * 3 + 1] - coord[c * 3 + 1];
        float dz = coord[r * 3 + 2] - coord[c * 3 + 2];
        sCD[tid][0] = dx; sCD[tid][1] = dy; sCD[tid][2] = dz;
        float rad = dx * dx + dy * dy + dz * dz;
        sIn[tid * 544 + 512] = (__bf16)rad;
        #pragma unroll
        for (int k = 513; k < 544; ++k) sIn[tid * 544 + k] = (__bf16)0.0f;
        atomicAdd(&cnt[r], 1.0f);
    }
    __syncthreads();

    // gather [h[row] | h[col]] -> LDS bf16
    for (int idx = tid; idx < 64 * 512; idx += 256) {
        int m = idx >> 9;
        int k = idx & 511;
        float v = (k < 256) ? h[(size_t)sR[m] * 256 + k]
                            : h[(size_t)sC[m] * 256 + (k - 256)];
        sIn[m * 544 + k] = (__bf16)v;
    }
    __syncthreads();

    const int wave  = tid >> 5;
    const int lane  = tid & 31;
    const int hi    = lane >> 4;
    const int ncol0 = wave * 16 + (lane & 15);
    const int ncol1 = ncol0 + 128;

    v8f acc[4][2];

    // edge-MLP layer 1: [64,544] x [544,256] (radial folded into K via col 512)
    zero_acc(acc);
    wave_gemm4x2<544>(sIn, 544, We1b, 544, ncol0, ncol1, lane, acc);
    #pragma unroll
    for (int s = 0; s < 2; ++s) {
        int ncol = s ? ncol1 : ncol0;
        float b = be1[ncol];
        #pragma unroll
        for (int mt = 0; mt < 4; ++mt)
            #pragma unroll
            for (int v = 0; v < 8; ++v)
                sM[(mt * 16 + v + hi * 8) * 256 + ncol] = (__bf16)silu_f(acc[mt][s][v] + b);
    }
    __syncthreads();

    // edge-MLP layer 2 -> edge_feat; scatter into agg (L2-resident atomics)
    zero_acc(acc);
    wave_gemm4x2<256>(sM, 256, We2b, 256, ncol0, ncol1, lane, acc);
    #pragma unroll
    for (int s = 0; s < 2; ++s) {
        int ncol = s ? ncol1 : ncol0;
        float b = be2[ncol];
        #pragma unroll
        for (int mt = 0; mt < 4; ++mt)
            #pragma unroll
            for (int v = 0; v < 8; ++v) {
                int m = mt * 16 + v + hi * 8;
                float f = silu_f(acc[mt][s][v] + b);
                sEF[m * 256 + ncol] = (__bf16)f;
                atomicAdd(&agg[(size_t)sR[m] * 256 + ncol], f);
            }
    }
    __syncthreads();

    // coord-MLP layer 1 (reuse sIn as output buffer)
    __bf16* sQ = sIn;
    zero_acc(acc);
    wave_gemm4x2<256>(sEF, 256, Wc1b, 256, ncol0, ncol1, lane, acc);
    #pragma unroll
    for (int s = 0; s < 2; ++s) {
        int ncol = s ? ncol1 : ncol0;
        float b = bc1[ncol];
        #pragma unroll
        for (int mt = 0; mt < 4; ++mt)
            #pragma unroll
            for (int v = 0; v < 8; ++v)
                sQ[(mt * 16 + v + hi * 8) * 256 + ncol] = (__bf16)silu_f(acc[mt][s][v] + b);
    }
    __syncthreads();

    // phi_x = q . Wc2 ; trans = coord_diff * phi_x ; scatter coord sums
    if (tid < 64) {
        float dot = 0.0f;
        #pragma unroll 8
        for (int k = 0; k < 256; ++k)
            dot += (float)sQ[tid * 256 + k] * Wc2[k];
        int r = sR[tid];
        atomicAdd(&cAgg[r * 3 + 0], sCD[tid][0] * dot);
        atomicAdd(&cAgg[r * 3 + 1], sCD[tid][1] * dot);
        atomicAdd(&cAgg[r * 3 + 2], sCD[tid][2] * dot);
    }
}

// ---------------- node kernel: one block = 64 nodes (tail-guarded) ----------------

__global__ void __launch_bounds__(256)
egnn_node_kernel(const float* __restrict__ h, const float* __restrict__ agg,
                 const __bf16* __restrict__ Wn1b, const float* __restrict__ bn1,
                 const __bf16* __restrict__ Wn2b, const float* __restrict__ bn2,
                 float* __restrict__ hout) {
    __shared__ __bf16 sIn[64 * 512];   // [h | agg]
    __shared__ __bf16 sM [64 * 256];

    const int tid = threadIdx.x;
    const int n0  = blockIdx.x * 64;

    for (int idx = tid; idx < 64 * 512; idx += 256) {
        int m = idx >> 9;
        int k = idx & 511;
        int node = n0 + m;
        if (node >= N_NODES) node = N_NODES - 1;   // clamp tail (stores are guarded)
        float v = (k < 256) ? h[(size_t)node * 256 + k]
                            : agg[(size_t)node * 256 + (k - 256)];
        sIn[m * 512 + k] = (__bf16)v;
    }
    __syncthreads();

    const int wave  = tid >> 5;
    const int lane  = tid & 31;
    const int hi    = lane >> 4;
    const int ncol0 = wave * 16 + (lane & 15);
    const int ncol1 = ncol0 + 128;

    v8f acc[4][2];

    zero_acc(acc);
    wave_gemm4x2<512>(sIn, 512, Wn1b, 512, ncol0, ncol1, lane, acc);
    #pragma unroll
    for (int s = 0; s < 2; ++s) {
        int ncol = s ? ncol1 : ncol0;
        float b = bn1[ncol];
        #pragma unroll
        for (int mt = 0; mt < 4; ++mt)
            #pragma unroll
            for (int v = 0; v < 8; ++v)
                sM[(mt * 16 + v + hi * 8) * 256 + ncol] = (__bf16)silu_f(acc[mt][s][v] + b);
    }
    __syncthreads();

    zero_acc(acc);
    wave_gemm4x2<256>(sM, 256, Wn2b, 256, ncol0, ncol1, lane, acc);
    #pragma unroll
    for (int s = 0; s < 2; ++s) {
        int ncol = s ? ncol1 : ncol0;
        float b = bn2[ncol];
        #pragma unroll
        for (int mt = 0; mt < 4; ++mt)
            #pragma unroll
            for (int v = 0; v < 8; ++v) {
                int node = n0 + mt * 16 + v + hi * 8;
                if (node < N_NODES)
                    hout[(size_t)node * 256 + ncol] =
                        acc[mt][s][v] + b + h[(size_t)node * 256 + ncol];  // residual
            }
    }
}

// ---------------- host launcher ----------------

extern "C" void kernel_launch(void* const* d_in, const int* in_sizes, int n_in,
                              void* d_out, int out_size, void* d_ws, size_t ws_size,
                              hipStream_t stream) {
    const float* h     = (const float*)d_in[0];
    const int*   ei    = (const int*)  d_in[1];
    const float* coord = (const float*)d_in[2];
    const float* We1 = (const float*)d_in[3];  const float* be1 = (const float*)d_in[4];
    const float* We2 = (const float*)d_in[5];  const float* be2 = (const float*)d_in[6];
    const float* Wn1 = (const float*)d_in[7];  const float* bn1 = (const float*)d_in[8];
    const float* Wn2 = (const float*)d_in[9];  const float* bn2 = (const float*)d_in[10];
    const float* Wc1 = (const float*)d_in[11]; const float* bc1 = (const float*)d_in[12];
    const float* Wc2 = (const float*)d_in[13];

    char* ws = (char*)d_ws;
    size_t off = 0;
    auto alloc = [&](size_t bytes) -> void* {
        void* p = ws + off;
        off = (off + bytes + 255) & ~(size_t)255;
        return p;
    };
    __bf16* We1b = (__bf16*)alloc((size_t)256 * 544 * 2);
    __bf16* We2b = (__bf16*)alloc((size_t)256 * 256 * 2);
    __bf16* Wc1b = (__bf16*)alloc((size_t)256 * 256 * 2);
    __bf16* Wn1b = (__bf16*)alloc((size_t)256 * 512 * 2);
    __bf16* Wn2b = (__bf16*)alloc((size_t)256 * 256 * 2);
    float*  agg  = (float*)alloc((size_t)N_NODES * 256 * 4);
    float*  cAgg = (float*)alloc((size_t)N_NODES * 3 * 4);
    float*  cnt  = (float*)alloc((size_t)N_NODES * 4);

    auto cvt = [&](const float* s, __bf16* d, int r, int c, int ld) {
        int tot = r * ld;
        convert_pad_kernel<<<(tot + 255) / 256, 256, 0, stream>>>(s, d, r, c, ld);
    };
    cvt(We1, We1b, 256, 513, 544);
    cvt(We2, We2b, 256, 256, 256);
    cvt(Wc1, Wc1b, 256, 256, 256);
    cvt(Wn1, Wn1b, 256, 512, 512);
    cvt(Wn2, Wn2b, 256, 256, 256);

    zero_kernel<<<(N_NODES * 256 + 255) / 256, 256, 0, stream>>>(agg,  N_NODES * 256);
    zero_kernel<<<(N_NODES * 3   + 255) / 256, 256, 0, stream>>>(cAgg, N_NODES * 3);
    zero_kernel<<<(N_NODES       + 255) / 256, 256, 0, stream>>>(cnt,  N_NODES);

    egnn_edge_kernel<<<N_EDGES / 64, 256, 0, stream>>>(
        h, ei, coord, We1b, be1, We2b, be2, Wc1b, bc1, Wc2, agg, cAgg, cnt);

    float* hout = (float*)d_out;
    float* cout = hout + (size_t)N_NODES * 256;
    egnn_node_kernel<<<(N_NODES + 63) / 64, 256, 0, stream>>>(
        h, agg, Wn1b, bn1, Wn2b, bn2, hout);
    coord_out_kernel<<<(N_NODES + 255) / 256, 256, 0, stream>>>(coord, cAgg, cnt, cout, N_NODES);
}
